// MambaVisionMixer_89936615178949
// MI455X (gfx1250) — compile-verified, matrix operand order checked by reading, not run on hardware
//
#include <hip/hip_runtime.h>
#include <hip/hip_bf16.h>
#include <math.h>

// ---------------------------------------------------------------------------
// Shapes (fixed by the reference)
// ---------------------------------------------------------------------------
#define D_MODEL 1024
#define D_INNER 2048
#define D_HALF  1024
#define D_STATE 16
#define DT_RANK 64
#define B_SZ    4
#define SEQ_L   2048
#define MROWS   (B_SZ * SEQ_L)          // 8192
#define XDBL_W  (DT_RANK + 2 * D_STATE) // 96

// ---------------------------------------------------------------------------
// WMMA / vector types (CDNA5, wave32)
// ---------------------------------------------------------------------------
typedef __attribute__((ext_vector_type(16))) __bf16 bf16x16;
typedef __attribute__((ext_vector_type(8)))  __bf16 bf16x8;
typedef __attribute__((ext_vector_type(4)))  __bf16 bf16x4;
typedef __attribute__((ext_vector_type(8)))  float  f32x8;
typedef __attribute__((ext_vector_type(4)))  float  f32x4;

union Frag {
    bf16x16 v;
    bf16x8  h[2];
};

// ---------------------------------------------------------------------------
// Tiled BF16 WMMA GEMM:  C(MxN) = A(MxK, f32) * B(KxN, f32)
// Block tile 128x128, 8 waves; each wave owns a 32x64 sub-tile = eight
// 16x16 WMMA tiles -> 8 v_wmma per K-chunk vs 12 ds_load_b128 + 2 barriers.
// K staged in LDS in chunks of 32 (bf16); B transposed in LDS so both A and
// B fragments are contiguous-K 16B ds loads (ISA 7.12.2 layouts).
// Requirements: M % 128 == 0, K % 32 == 0, lda/ldb % 4 == 0 (true for all
// four shapes here). GUARD adds N-bounds checks (only the N=96 projection).
//
// MODE 0: C = acc
// MODE 2: C = softplus(acc + 2*aux[col])   (reference adds inv_dt twice)
// MODE 3: C = acc + aux[col]
// ---------------------------------------------------------------------------
#define BM 128
#define BN 128
#define KB 32

template<int MODE, bool GUARD>
__global__ __launch_bounds__(256)
void gemm_bf16_wmma(const float* __restrict__ A, int lda,
                    const float* __restrict__ B, int ldb,
                    float* __restrict__ C, int ldc,
                    const float* __restrict__ aux,
                    int M, int N, int K)
{
    __shared__ alignas(16) __bf16 As[BM][KB];
    __shared__ alignas(16) __bf16 Bs[BN][KB];   // B^T: Bs[n][k]

    const int tid  = threadIdx.x;
    const int wave = tid >> 5;
    const int lane = tid & 31;
    const int m0   = blockIdx.y * BM;
    const int n0   = blockIdx.x * BN;

    // 8 waves -> 4 along M (32 rows each) x 2 along N (64 cols each)
    const int wm = (wave & 3) * 32;
    const int wn = (wave >> 2) * 64;

    const int hh = lane >> 4;         // lane half selects K phase
    const int ml = lane & 15;         // row/col within a 16x16 tile

    f32x8 acc[2][4];
    #pragma unroll
    for (int i = 0; i < 2; ++i)
        #pragma unroll
        for (int j = 0; j < 4; ++j)
            acc[i][j] = f32x8{};

    // Cooperative-load indexing (hoisted; branch-free inner loop)
    // A tile: BM x KB = 4096 f32 = 1024 float4; 4 float4 per thread.
    //   idx4 -> row = idx4>>3, kq = idx4&7 (k = 4*kq)
    // B tile: KB x BN = 1024 float4;  idx4 -> kk = idx4>>5, nq = idx4&31
    for (int k0 = 0; k0 < K; k0 += KB) {
        #pragma unroll
        for (int i = 0; i < 4; ++i) {
            int idx4 = tid + i * 256;
            int r    = idx4 >> 3;
            int kq   = idx4 & 7;
            const float* p = A + (size_t)(m0 + r) * lda + (k0 + kq * 4);
            f32x4 v = *(const f32x4*)p;
            bf16x4 w;
            #pragma unroll
            for (int e = 0; e < 4; ++e) w[e] = (__bf16)v[e];
            *(bf16x4*)&As[r][kq * 4] = w;      // 8B ds store
        }
        #pragma unroll
        for (int i = 0; i < 4; ++i) {
            int idx4 = tid + i * 256;
            int kk   = idx4 >> 5;
            int nq   = idx4 & 31;              // consecutive tid -> coalesced
            int gn   = n0 + nq * 4;
            f32x4 v = f32x4{};
            if (!GUARD || gn < N)              // N%4==0 so whole-vec guard ok
                v = *(const f32x4*)(B + (size_t)(k0 + kk) * ldb + gn);
            #pragma unroll
            for (int e = 0; e < 4; ++e)
                Bs[nq * 4 + e][kk] = (__bf16)v[e];   // transpose scatter
        }
        __syncthreads();

        // Fragment loads: two 16B ds reads each (contiguous K)
        Frag a[2], b[4];
        #pragma unroll
        for (int i = 0; i < 2; ++i) {
            const __bf16* p = &As[wm + i * 16 + ml][hh * 8];
            a[i].h[0] = *(const bf16x8*)p;
            a[i].h[1] = *(const bf16x8*)(p + 16);
        }
        #pragma unroll
        for (int j = 0; j < 4; ++j) {
            const __bf16* p = &Bs[wn + j * 16 + ml][hh * 8];
            b[j].h[0] = *(const bf16x8*)p;
            b[j].h[1] = *(const bf16x8*)(p + 16);
        }

        #pragma unroll
        for (int i = 0; i < 2; ++i)
            #pragma unroll
            for (int j = 0; j < 4; ++j)
                acc[i][j] = __builtin_amdgcn_wmma_f32_16x16x32_bf16(
                    false, a[i].v, false, b[j].v, (short)0, acc[i][j],
                    false, false);

        __syncthreads();
    }

    // Epilogue. D layout: lane(0-15) N=lane, M=vgpr; lane(16-31) M=vgpr+8.
    #pragma unroll
    for (int i = 0; i < 2; ++i) {
        const int m_base = m0 + wm + i * 16 + hh * 8;
        #pragma unroll
        for (int j = 0; j < 4; ++j) {
            const int gn = n0 + wn + j * 16 + ml;
            if (GUARD && gn >= N) continue;
            float avals = 0.0f;
            if (MODE != 0) avals = aux[gn];
            #pragma unroll
            for (int v = 0; v < 8; ++v) {
                const int gm = m_base + v;
                float r = acc[i][j][v];
                float o;
                if (MODE == 2) {
                    float x = r + 2.0f * avals;
                    o = (x > 20.0f) ? x : log1pf(__expf(x));
                } else if (MODE == 3) {
                    o = r + avals;
                } else {
                    o = r;
                }
                C[(size_t)gm * ldc + gn] = o;
            }
        }
    }
}

// ---------------------------------------------------------------------------
// Depthwise conv (k=4, SAME: pad_lo=1, pad_hi=2, cross-correlation) + SiLU.
// Reads x (row stride istride, already column-offset) and writes y (row
// stride ostride, already column-offset). Rows are (b*SEQ_L + t).
// ---------------------------------------------------------------------------
__global__ __launch_bounds__(256)
void dwconv_silu_kernel(const float* __restrict__ x, int istride,
                        const float* __restrict__ w,
                        const float* __restrict__ bias,
                        float* __restrict__ y, int ostride,
                        long long total)
{
    long long i = (long long)blockIdx.x * blockDim.x + threadIdx.x;
    if (i >= total) return;
    int c   = (int)(i % D_HALF);
    int row = (int)(i / D_HALF);       // 0..MROWS-1
    int t   = row & (SEQ_L - 1);
    int b   = row >> 11;               // SEQ_L == 2048

    float s = bias[c];
    #pragma unroll
    for (int j = 0; j < 4; ++j) {
        int tt = t + j - 1;
        if (tt >= 0 && tt < SEQ_L)
            s = fmaf(x[(size_t)(b * SEQ_L + tt) * istride + c],
                     w[j * D_HALF + c], s);
    }
    y[(size_t)row * ostride + c] = s / (1.0f + __expf(-s));   // SiLU
}

// ---------------------------------------------------------------------------
// Selective scan: one lane per (b, d); 16 states in registers.
// A[d][n] = -(n+1)  =>  dA_n = exp(-delta)^(n+1): one exp + running power.
// B_t / C_t (32 floats, shared across all d of one b) staged through LDS.
// Writes y with row stride ystride (into the concat buffer, cols 0..1023).
// ---------------------------------------------------------------------------
__global__ __launch_bounds__(256)
void selective_scan_kernel(const float* __restrict__ delta,
                           const float* __restrict__ u,
                           const float* __restrict__ xdbl,
                           const float* __restrict__ Dvec,
                           float* __restrict__ y, int ystride)
{
    const int gid = blockIdx.x * 256 + threadIdx.x;   // 0..4095
    const int b   = gid >> 10;        // whole block shares b (1024 % 256 == 0)
    const int d   = gid & (D_HALF - 1);

    __shared__ float bc[2 * D_STATE];

    float h[D_STATE];
    #pragma unroll
    for (int n = 0; n < D_STATE; ++n) h[n] = 0.0f;

    const float dval = Dvec[d];

    for (int t = 0; t < SEQ_L; ++t) {
        const long long row = (long long)b * SEQ_L + t;
        if (threadIdx.x < 2 * D_STATE)
            bc[threadIdx.x] = xdbl[row * XDBL_W + DT_RANK + threadIdx.x];
        __syncthreads();

        float dlt = delta[row * D_HALF + d];
        float uu  = u[row * D_HALF + d];
        float r   = __expf(-dlt);       // dA_n = r^(n+1)
        float du  = dlt * uu;

        float p = 1.0f, acc = 0.0f;
        #pragma unroll
        for (int n = 0; n < D_STATE; ++n) {
            p *= r;
            h[n] = fmaf(p, h[n], du * bc[n]);
            acc  = fmaf(h[n], bc[D_STATE + n], acc);
        }
        y[row * ystride + d] = fmaf(uu, dval, acc);
        __syncthreads();
    }
}

// ---------------------------------------------------------------------------
// Pipeline:
//   1) xz = x @ W_in                        (WMMA, plain)       -> xz
//   2) conv+SiLU on xz[:, :1024] -> xs_c ;  xz[:, 1024:] -> ycat[:, 1024:]
//   3) x_dbl = xs_c @ W_xdbl                (WMMA, guarded N=96)
//   4) delta = softplus(xdbl[:, :64] @ W_dt + 2*inv_dt)  (WMMA fused)
//   5) selective scan -> ycat[:, :1024]
//   6) out = ycat @ W_out + b_out           (WMMA fused bias)
// ---------------------------------------------------------------------------
extern "C" void kernel_launch(void* const* d_in, const int* in_sizes, int n_in,
                              void* d_out, int out_size, void* d_ws, size_t ws_size,
                              hipStream_t stream)
{
    (void)in_sizes; (void)n_in; (void)out_size; (void)ws_size;

    const float* x       = (const float*)d_in[0];
    const float* W_in    = (const float*)d_in[1];
    const float* conv_xw = (const float*)d_in[2];
    const float* conv_xb = (const float*)d_in[3];
    const float* conv_zw = (const float*)d_in[4];
    const float* conv_zb = (const float*)d_in[5];
    const float* W_xdbl  = (const float*)d_in[6];
    const float* W_dt    = (const float*)d_in[7];
    const float* inv_dt  = (const float*)d_in[8];
    const float* Dvec    = (const float*)d_in[9];
    const float* W_out   = (const float*)d_in[10];
    const float* b_out   = (const float*)d_in[11];
    float* out = (float*)d_out;

    // Workspace (floats): xz (reused as delta), xs_c, ycat, xdbl
    float* ws   = (float*)d_ws;
    const size_t PLANE2 = (size_t)MROWS * D_INNER;    // 8192*2048
    const size_t PLANE1 = (size_t)MROWS * D_HALF;     // 8192*1024
    float* xz    = ws;                                // PLANE2
    float* xs_c  = xz + PLANE2;                       // PLANE1
    float* ycat  = xs_c + PLANE1;                     // PLANE2 (y | z_c)
    float* xdbl  = ycat + PLANE2;                     // MROWS * 96
    float* delta = xz;                                // reuse after convs

    dim3 blk(256);

    // 1) xz = x @ W_in
    gemm_bf16_wmma<0, false><<<dim3(D_INNER / BN, MROWS / BM), blk, 0, stream>>>(
        x, D_MODEL, W_in, D_INNER, xz, D_INNER, nullptr,
        MROWS, D_INNER, D_MODEL);

    // 2) depthwise conv + SiLU on both halves of xz
    {
        long long total = (long long)MROWS * D_HALF;
        dim3 grid((unsigned)((total + 255) / 256));
        dwconv_silu_kernel<<<grid, blk, 0, stream>>>(
            xz, D_INNER, conv_xw, conv_xb, xs_c, D_HALF, total);
        dwconv_silu_kernel<<<grid, blk, 0, stream>>>(
            xz + D_HALF, D_INNER, conv_zw, conv_zb, ycat + D_HALF, D_INNER, total);
    }

    // 3) x_dbl = xs_c @ W_xdbl   (N = 96 -> guarded)
    gemm_bf16_wmma<0, true><<<dim3(1, MROWS / BM), blk, 0, stream>>>(
        xs_c, D_HALF, W_xdbl, XDBL_W, xdbl, XDBL_W, nullptr,
        MROWS, XDBL_W, D_HALF);

    // 4) delta = softplus(xdbl[:, :64] @ W_dt + 2*inv_dt)
    gemm_bf16_wmma<2, false><<<dim3(D_HALF / BN, MROWS / BM), blk, 0, stream>>>(
        xdbl, XDBL_W, W_dt, D_HALF, delta, D_HALF, inv_dt,
        MROWS, D_HALF, DT_RANK);

    // 5) selective scan -> ycat[:, :1024]
    selective_scan_kernel<<<dim3((B_SZ * D_HALF) / 256), blk, 0, stream>>>(
        delta, xs_c, xdbl, Dvec, ycat, D_INNER);

    // 6) out = ycat @ W_out + b_out
    gemm_bf16_wmma<3, false><<<dim3(D_MODEL / BN, MROWS / BM), blk, 0, stream>>>(
        ycat, D_INNER, W_out, D_MODEL, out, D_MODEL, b_out,
        MROWS, D_MODEL, D_INNER);
}
